// GRUModelForConditionalGeneration_36610301231683
// MI455X (gfx1250) — compile-verified, matrix operand order checked by reading, not run on hardware
//
#include <hip/hip_runtime.h>
#include <hip/hip_bf16.h>
#include <math.h>

// ---- model dims ----
#define DV 32000
#define DE 1024
#define DH 512
#define DL 2
#define DB 32
#define DS 128
#define DTL 63
#define DH2 1024      // 2H
#define DH3 1536      // 3H
#define DKDEC0 2048   // E + 2H
#define DKOUT 2560    // H + 2H + E
#define DTOUT 64      // TL + 1
#define FNEG (-1e10f)

typedef __attribute__((ext_vector_type(16))) _Float16 v16h;
typedef __attribute__((ext_vector_type(8)))  _Float16 v8h;
typedef __attribute__((ext_vector_type(8)))  float    v8f;

// ---------------------------------------------------------------------------
// WMMA 16x16x32 f16 fragment helpers (wave32).
// A-fragment (16x32, MxK): lanes 0-15 row M=lane, K {k0..k0+7, k0+16..k0+23};
// lanes 16-31 row M=lane-16, K {k0+8..k0+15, k0+24..k0+31}. Both halves are
// contiguous 8-half runs -> two 16B loads.
// B operand for C = A * W^T with W stored (N,K) row-major uses the identical
// loader with "row" = n (rows of W are columns of W^T).
// ---------------------------------------------------------------------------
__device__ __forceinline__ v16h frag_ld(const _Float16* base, int ld, int row0, int k0) {
  const int lane = threadIdx.x & 31;
  const _Float16* p = base + (size_t)(row0 + (lane & 15)) * ld + k0 + ((lane >> 4) << 3);
  v8h lo = *reinterpret_cast<const v8h*>(p);
  v8h hi = *reinterpret_cast<const v8h*>(p + 16);
  v16h f;
#pragma unroll
  for (int e = 0; e < 8; ++e) { f[e] = lo[e]; f[e + 8] = hi[e]; }
  return f;
}

__device__ __forceinline__ v8f wmma_step(v16h a, v16h b, v8f c) {
  return __builtin_amdgcn_wmma_f32_16x16x32_f16(false, a, false, b, (short)0,
                                                c, false, false);
}

// C/D fragment: VGPR r -> M = r + 8*(lane>=16), N = lane&15.
__device__ __forceinline__ void frag_st(float* C, long ldc, int row0, int col0,
                                        v8f acc, const float* bias, int act) {
  const int lane = threadIdx.x & 31;
  const int n  = col0 + (lane & 15);
  const int m0 = row0 + ((lane >> 4) << 3);
  const float bv = bias ? bias[n] : 0.f;
#pragma unroll
  for (int r = 0; r < 8; ++r) {
    float v = acc[r] + bv;
    if (act == 1) v = tanhf(v);
    C[(size_t)(m0 + r) * ldc + n] = v;
  }
}

// Generic GEMM: C(M,N) f32 = A(M,K) f16 * W(N,K)^T f16 (+bias, optional tanh).
// One wave per 32x16 tile (2 M-tiles share each W fragment -> 2 WMMA per W
// load, halving the dominant weight stream), 4 waves/block along N.
// Requires N%64==0, M%32==0, K%32==0.
__global__ void k_gemm(const _Float16* __restrict__ A, int lda,
                       const _Float16* __restrict__ W, int ldw,
                       const float* __restrict__ bias,
                       float* __restrict__ C, long ldc, int K, int act) {
  const int n0 = (blockIdx.x * (blockDim.x >> 5) + (threadIdx.x >> 5)) << 4;
  const int m0 = blockIdx.y << 5;
  v8f acc0 = {}, acc1 = {};
  for (int k0 = 0; k0 < K; k0 += 32) {
    v16h b  = frag_ld(W, ldw, n0, k0);
    v16h a0 = frag_ld(A, lda, m0, k0);
    v16h a1 = frag_ld(A, lda, m0 + 16, k0);
    acc0 = wmma_step(a0, b, acc0);
    acc1 = wmma_step(a1, b, acc1);
  }
  frag_st(C, ldc, m0,      n0, acc0, bias, act);
  frag_st(C, ldc, m0 + 16, n0, acc1, bias, act);
}

// ---------------------------------------------------------------------------
// Persistent encoder GRU direction: 1 workgroup, 32 wave32. h (f16) lives in
// LDS; each step does gh = h@Whh^T + bhh via WMMA (96 n-tiles x 2 m-tiles over
// 32 waves, 3 per wave), then the fused GRU nonlinearity with the precomputed
// gi = x@Wih^T + bih.
// ---------------------------------------------------------------------------
__global__ __launch_bounds__(1024) void k_gru_seq(
    const float* __restrict__ gi,      // (S,B,3H) includes bih
    const _Float16* __restrict__ Whh,  // (3H,H) f16
    const float* __restrict__ bhh,     // (3H)
    const int* __restrict__ amask,     // (B,S)
    float* __restrict__ outs,          // (S,B,H)
    float* __restrict__ hT,            // (B,H) running + final hidden (f32)
    float* __restrict__ ghb,           // (B,3H) scratch
    int reverse) {
  __shared__ __align__(16) _Float16 hsh[DB * DH];   // 32 KB
  const int tid = threadIdx.x;
  for (int i = tid; i < DB * DH; i += 1024) { hsh[i] = (_Float16)0.f; hT[i] = 0.f; }
  __syncthreads();
  const int wave = tid >> 5;
#pragma unroll 1
  for (int ti = 0; ti < DS; ++ti) {
    const int t = reverse ? (DS - 1 - ti) : ti;
#pragma unroll 1
    for (int i = 0; i < 3; ++i) {                 // 96 n-tiles / 32 waves
      const int n0 = (wave + (i << 5)) << 4;
      v8f acc0 = {}, acc1 = {};
#pragma unroll 1
      for (int k0 = 0; k0 < DH; k0 += 32) {
        v16h b  = frag_ld(Whh, DH, n0, k0);
        v16h a0 = frag_ld(hsh, DH, 0,  k0);
        v16h a1 = frag_ld(hsh, DH, 16, k0);
        acc0 = wmma_step(a0, b, acc0);
        acc1 = wmma_step(a1, b, acc1);
      }
      frag_st(ghb, DH3, 0,  n0, acc0, bhh, 0);
      frag_st(ghb, DH3, 16, n0, acc1, bhh, 0);
    }
    __threadfence();
    __syncthreads();
#pragma unroll 1
    for (int e = 0; e < 16; ++e) {                // B*H = 16384 elems / 1024 thr
      const int id = tid + (e << 10);
      const int b = id >> 9, j = id & 511;
      const float* gir = gi + (size_t)(t * DB + b) * DH3;
      const float* ghr = ghb + (size_t)b * DH3;
      const float r  = 1.f / (1.f + expf(-(gir[j] + ghr[j])));
      const float z  = 1.f / (1.f + expf(-(gir[DH + j] + ghr[DH + j])));
      const float n  = tanhf(gir[DH2 + j] + r * ghr[DH2 + j]);
      const float ho = hT[id];
      const float hn = (1.f - z) * n + z * ho;
      const bool  m  = amask[b * DS + t] != 0;
      const float hm = m ? hn : ho;
      outs[(size_t)(t * DB + b) * DH + j] = m ? hn : 0.f;
      hT[id] = hm;
      hsh[id] = (_Float16)hm;
    }
    __syncthreads();
  }
}

// ---------------------------------------------------------------------------
// Small elementwise / packing kernels
// ---------------------------------------------------------------------------
__global__ void k_cvt(const float* __restrict__ a, _Float16* __restrict__ o, long n) {
  long i = (long)blockIdx.x * blockDim.x + threadIdx.x;
  const long st = (long)gridDim.x * blockDim.x;
  for (; i < n; i += st) o[i] = (_Float16)a[i];
}

__global__ void k_split_attn(const float* __restrict__ w,
                             _Float16* __restrict__ w1, _Float16* __restrict__ w2) {
  const int id = blockIdx.x * blockDim.x + threadIdx.x;   // H*3H
  if (id >= DH * DH3) return;
  const int i = id / DH3, j = id % DH3;
  const float v = w[id];
  if (j < DH) w1[i * DH + j] = (_Float16)v;
  else        w2[i * DH2 + (j - DH)] = (_Float16)v;
}

__global__ void k_enc_embed(const float* __restrict__ emb, const int* __restrict__ ids,
                            _Float16* __restrict__ x) {
  const long id = (long)blockIdx.x * blockDim.x + threadIdx.x;  // S*B*E
  if (id >= (long)DS * DB * DE) return;
  const long row = id / DE; const int e = (int)(id % DE);
  const int t = (int)(row / DB), b = (int)(row % DB);
  x[id] = (_Float16)emb[(size_t)ids[b * DS + t] * DE + e];
}

__global__ void k_concat2(const float* __restrict__ a, const float* __restrict__ b,
                          long rows, _Float16* __restrict__ oh, float* __restrict__ of) {
  const long id = (long)blockIdx.x * blockDim.x + threadIdx.x;  // rows*2H
  if (id >= rows * DH2) return;
  const long r = id / DH2; const int c = (int)(id % DH2);
  const float v = (c < DH) ? a[r * DH + c] : b[r * DH + (c - DH)];
  oh[id] = (_Float16)v;
  if (of) of[id] = v;
}

__global__ void k_tokens(const int* __restrict__ labels, int* __restrict__ tok) {
  const int id = blockIdx.x * blockDim.x + threadIdx.x;   // TL*B
  if (id >= DTL * DB) return;
  const int t = id / DB, b = id % DB;
  int v;
  if (t == 0) v = 1;                               // START
  else { v = labels[b * DTL + (t - 1)]; if (v == -100) v = 0; }  // PAD
  tok[id] = v;
}

__global__ void k_zero0(float* __restrict__ out) {
  const long id = (long)blockIdx.x * blockDim.x + threadIdx.x;  // B*V
  if (id >= (long)DB * DV) return;
  const long b = id / DV, v = id % DV;
  out[b * (long)DTOUT * DV + v] = 0.f;
}

__global__ void k_dec_emb(const float* __restrict__ emb, const int* __restrict__ tok_t,
                          _Float16* __restrict__ xl, _Float16* __restrict__ cat) {
  const int id = blockIdx.x * blockDim.x + threadIdx.x;   // B*E
  if (id >= DB * DE) return;
  const int b = id / DE, e = id % DE;
  const float v = emb[(size_t)tok_t[b] * DE + e];
  xl[b * DKDEC0 + e] = (_Float16)v;                 // xl = [emb, weighted]
  cat[b * DKOUT + DH3 + e] = (_Float16)v;           // cat = [h_top, weighted, emb]
}

// Attention: scores = tanh(hproj + encproj + b) . v ; masked softmax over S;
// weighted = a @ enc ; writes weighted into f32 buf + f16 xl/cat slots.
__global__ void k_attn(const float* __restrict__ hproj,    // (B,H)
                       const float* __restrict__ encproj,  // (S*B,H)
                       const float* __restrict__ attn_b,   // (H)
                       const float* __restrict__ attn_v,   // (H)
                       const int* __restrict__ amask,      // (B,S)
                       const float* __restrict__ enc,      // (S,B,2H) f32
                       float* __restrict__ wsum,           // (B,2H)
                       _Float16* __restrict__ xl, _Float16* __restrict__ cat) {
  const int b = blockIdx.x, s = threadIdx.x;     // 32 blocks x 128 threads
  __shared__ float sc[DS];
  float e = FNEG;
  if (amask[b * DS + s]) {
    float acc = 0.f;
    const float* hp = hproj + b * DH;
    const float* ep = encproj + (size_t)(s * DB + b) * DH;
    for (int j = 0; j < DH; ++j)
      acc += attn_v[j] * tanhf(hp[j] + ep[j] + attn_b[j]);
    e = acc;
  }
  sc[s] = e; __syncthreads();
  float mx = -INFINITY;
  for (int i = 0; i < DS; ++i) mx = fmaxf(mx, sc[i]);
  __syncthreads();
  sc[s] = expf(e - mx); __syncthreads();
  float sum = 0.f;
  for (int i = 0; i < DS; ++i) sum += sc[i];
  const float inv = 1.f / sum;
  for (int d = s; d < DH2; d += DS) {
    float w = 0.f;
    for (int i = 0; i < DS; ++i) w += sc[i] * enc[(size_t)(i * DB + b) * DH2 + d];
    w *= inv;
    wsum[b * DH2 + d] = w;
    xl[b * DKDEC0 + DE + d] = (_Float16)w;
    cat[b * DKOUT + DH + d] = (_Float16)w;
  }
}

// Decoder GRU combine: h' = (1-z)*n + z*h ; updates f32 + f16 h, optional cat.
__global__ void k_dec_comb(const float* __restrict__ gi, const float* __restrict__ gh,
                           float* __restrict__ hf, _Float16* __restrict__ hh,
                           _Float16* __restrict__ cat) {
  const int id = blockIdx.x * blockDim.x + threadIdx.x;   // B*H
  if (id >= DB * DH) return;
  const int b = id >> 9, j = id & 511;
  const float* gir = gi + (size_t)b * DH3;
  const float* ghr = gh + (size_t)b * DH3;
  const float r = 1.f / (1.f + expf(-(gir[j] + ghr[j])));
  const float z = 1.f / (1.f + expf(-(gir[DH + j] + ghr[DH + j])));
  const float n = tanhf(gir[DH2 + j] + r * ghr[DH2 + j]);
  const float h = (1.f - z) * n + z * hf[id];
  hf[id] = h;
  hh[id] = (_Float16)h;
  if (cat) cat[b * DKOUT + j] = (_Float16)h;
}

// ---------------------------------------------------------------------------
extern "C" void kernel_launch(void* const* d_in, const int* in_sizes, int n_in,
                              void* d_out, int out_size, void* d_ws, size_t ws_size,
                              hipStream_t stream) {
  (void)in_sizes; (void)n_in; (void)out_size; (void)ws_size;
  const int*   input_ids = (const int*)  d_in[0];
  const int*   amask     = (const int*)  d_in[1];
  const int*   labels    = (const int*)  d_in[2];
  const float* enc_emb   = (const float*)d_in[3];
  const float* enc_Wih   = (const float*)d_in[4];
  const float* enc_Whh   = (const float*)d_in[5];
  const float* enc_bih   = (const float*)d_in[6];
  const float* enc_bhh   = (const float*)d_in[7];
  const float* fc_W      = (const float*)d_in[8];
  const float* fc_b      = (const float*)d_in[9];
  const float* attn_W    = (const float*)d_in[10];
  const float* attn_b    = (const float*)d_in[11];
  const float* attn_v    = (const float*)d_in[12];
  const float* dec_emb   = (const float*)d_in[13];
  const float* dec_Wih0  = (const float*)d_in[14];
  const float* dec_Wihr  = (const float*)d_in[15];
  const float* dec_Whh   = (const float*)d_in[16];
  const float* dec_bih   = (const float*)d_in[17];
  const float* dec_bhh   = (const float*)d_in[18];
  const float* out_W     = (const float*)d_in[19];
  const float* out_b     = (const float*)d_in[20];
  float* out = (float*)d_out;

  size_t off = 0;
  auto alloc = [&](size_t bytes) -> void* {
    void* p = (char*)d_ws + off;
    off += (bytes + 255) & ~(size_t)255;
    return p;
  };
  // f16 weight copies (out_W f16 = 164MB -> fits MI455X 192MB L2 across 63 reuses)
  _Float16* encWih_h = (_Float16*)alloc((size_t)DL*2*DH3*DH2*2);
  _Float16* encWhh_h = (_Float16*)alloc((size_t)DL*2*DH3*DH*2);
  _Float16* fcW_h    = (_Float16*)alloc((size_t)DL*DH*DH2*2);
  _Float16* aW1_h    = (_Float16*)alloc((size_t)DH*DH*2);
  _Float16* aW2_h    = (_Float16*)alloc((size_t)DH*DH2*2);
  _Float16* dWih0_h  = (_Float16*)alloc((size_t)DH3*DKDEC0*2);
  _Float16* dWihr_h  = (_Float16*)alloc((size_t)DH3*DH*2);
  _Float16* dWhh_h   = (_Float16*)alloc((size_t)DL*DH3*DH*2);
  _Float16* outW_h   = (_Float16*)alloc((size_t)DV*DKOUT*2);
  // activations
  _Float16* x0_h   = (_Float16*)alloc((size_t)DS*DB*DE*2);
  _Float16* x2_h   = (_Float16*)alloc((size_t)DS*DB*DH2*2);
  float*    enc_f  = (float*)   alloc((size_t)DS*DB*DH2*4);
  float*    giF    = (float*)   alloc((size_t)DS*DB*DH3*4);
  float*    giB    = (float*)   alloc((size_t)DS*DB*DH3*4);
  float*    fo     = (float*)   alloc((size_t)DS*DB*DH*4);
  float*    bo     = (float*)   alloc((size_t)DS*DB*DH*4);
  float*    hTf    = (float*)   alloc((size_t)DB*DH*4);
  float*    hTb    = (float*)   alloc((size_t)DB*DH*4);
  float*    ghF    = (float*)   alloc((size_t)DB*DH3*4);
  float*    ghB    = (float*)   alloc((size_t)DB*DH3*4);
  float*    encprj = (float*)   alloc((size_t)DS*DB*DH*4);
  _Float16* fcin_h = (_Float16*)alloc((size_t)DB*DH2*2);
  float*    hid_f  = (float*)   alloc((size_t)DL*DB*DH*4);
  _Float16* hid_h  = (_Float16*)alloc((size_t)DL*DB*DH*2);
  int*      tok    = (int*)     alloc((size_t)DTL*DB*4);
  _Float16* xl_h   = (_Float16*)alloc((size_t)DB*DKDEC0*2);
  _Float16* cat_h  = (_Float16*)alloc((size_t)DB*DKOUT*2);
  float*    hproj  = (float*)   alloc((size_t)DB*DH*4);
  float*    wsum   = (float*)   alloc((size_t)DB*DH2*4);
  float*    gid    = (float*)   alloc((size_t)DB*DH3*4);
  float*    ghd    = (float*)   alloc((size_t)DB*DH3*4);

  const int TPB = 256;
  auto cvt = [&](const float* a, _Float16* o, long n) {
    int blocks = (int)((n + TPB - 1) / TPB);
    if (blocks > 262144) blocks = 262144;           // grid-stride
    k_cvt<<<blocks, TPB, 0, stream>>>(a, o, n);
  };
  cvt(enc_Wih,  encWih_h, (long)DL*2*DH3*DH2);
  cvt(enc_Whh,  encWhh_h, (long)DL*2*DH3*DH);
  cvt(fc_W,     fcW_h,    (long)DL*DH*DH2);
  cvt(dec_Wih0, dWih0_h,  (long)DH3*DKDEC0);
  cvt(dec_Wihr, dWihr_h,  (long)DH3*DH);
  cvt(dec_Whh,  dWhh_h,   (long)DL*DH3*DH);
  cvt(out_W,    outW_h,   (long)DV*DKOUT);
  k_split_attn<<<(DH*DH3 + TPB - 1)/TPB, TPB, 0, stream>>>(attn_W, aW1_h, aW2_h);

  { long n = (long)DS*DB*DE;
    k_enc_embed<<<(int)((n + TPB - 1)/TPB), TPB, 0, stream>>>(enc_emb, input_ids, x0_h); }

  auto gemm = [&](const _Float16* A, const _Float16* W, const float* bias,
                  float* C, long ldc, int M, int N, int K, int act) {
    dim3 g(N / 64, M / 32), b(128);
    k_gemm<<<g, b, 0, stream>>>(A, K, W, K, bias, C, ldc, K, act);
  };

  // ---------------- encoder ----------------
  const _Float16* x = x0_h;
  for (int l = 0; l < DL; ++l) {
    gemm(x, encWih_h + (size_t)(l*2+0)*DH3*DH2, enc_bih + (l*2+0)*DH3, giF, DH3, DS*DB, DH3, DH2, 0);
    gemm(x, encWih_h + (size_t)(l*2+1)*DH3*DH2, enc_bih + (l*2+1)*DH3, giB, DH3, DS*DB, DH3, DH2, 0);
    k_gru_seq<<<1, 1024, 0, stream>>>(giF, encWhh_h + (size_t)(l*2+0)*DH3*DH,
                                      enc_bhh + (l*2+0)*DH3, amask, fo, hTf, ghF, 0);
    k_gru_seq<<<1, 1024, 0, stream>>>(giB, encWhh_h + (size_t)(l*2+1)*DH3*DH,
                                      enc_bhh + (l*2+1)*DH3, amask, bo, hTb, ghB, 1);
    long n = (long)DS*DB*DH2;
    k_concat2<<<(int)((n + TPB - 1)/TPB), TPB, 0, stream>>>(fo, bo, (long)DS*DB, x2_h,
                                                            (l == DL-1) ? enc_f : (float*)nullptr);
    x = x2_h;
  }
  // step-invariant attention projection of enc_out (hoists 400 GFLOP out of the loop)
  gemm(x2_h, aW2_h, nullptr, encprj, DH, DS*DB, DH, DH2, 0);
  // decoder hidden init: hid[l] = tanh([ff,bf] @ fc_W[l]^T + fc_b[l])
  { long n = (long)DB*DH2;
    k_concat2<<<(int)((n + TPB - 1)/TPB), TPB, 0, stream>>>(hTf, hTb, (long)DB, fcin_h, (float*)nullptr); }
  for (int l = 0; l < DL; ++l)
    gemm(fcin_h, fcW_h + (size_t)l*DH*DH2, fc_b + l*DH, hid_f + (size_t)l*DB*DH, DH, DB, DH, DH2, 1);
  cvt(hid_f, hid_h, (long)DL*DB*DH);
  k_tokens<<<(DTL*DB + TPB - 1)/TPB, TPB, 0, stream>>>(labels, tok);
  { long n = (long)DB*DV;
    k_zero0<<<(int)((n + TPB - 1)/TPB), TPB, 0, stream>>>(out); }

  // ---------------- decoder: 63 steps ----------------
  for (int t = 0; t < DTL; ++t) {
    k_dec_emb<<<(DB*DE + TPB - 1)/TPB, TPB, 0, stream>>>(dec_emb, tok + t*DB, xl_h, cat_h);
    gemm(hid_h + (size_t)(DL-1)*DB*DH, aW1_h, nullptr, hproj, DH, DB, DH, DH, 0);
    k_attn<<<DB, DS, 0, stream>>>(hproj, encprj, attn_b, attn_v, amask, enc_f,
                                  wsum, xl_h, cat_h);
    // layer 0: x = [emb, weighted]
    gemm(xl_h, dWih0_h, dec_bih, gid, DH3, DB, DH3, DKDEC0, 0);
    gemm(hid_h, dWhh_h, dec_bhh, ghd, DH3, DB, DH3, DH, 0);
    k_dec_comb<<<(DB*DH + TPB - 1)/TPB, TPB, 0, stream>>>(gid, ghd, hid_f, hid_h,
                                                          (_Float16*)nullptr);
    // layer 1: x = h0_new
    gemm(hid_h, dWihr_h, dec_bih + DH3, gid, DH3, DB, DH3, DH, 0);
    gemm(hid_h + (size_t)DB*DH, dWhh_h + (size_t)DH3*DH, dec_bhh + DH3, ghd, DH3, DB, DH3, DH, 0);
    k_dec_comb<<<(DB*DH + TPB - 1)/TPB, TPB, 0, stream>>>(gid, ghd, hid_f + (size_t)DB*DH,
                                                          hid_h + (size_t)DB*DH, cat_h);
    // logits[:, t+1, :] = cat @ out_W^T + out_b   (out_W f16 resident in L2)
    gemm(cat_h, outW_h, out_b, out + (size_t)(t + 1)*DV, (long)DTOUT*DV, DB, DV, DKOUT, 0);
  }
}
